// GptOssAttention_11424613007752
// MI455X (gfx1250) — matrix operand order repaired and checked by reading, not compiled, not run on hardware
//
#include <hip/hip_runtime.h>
#include <hip/hip_bf16.h>

// ---- problem constants (match reference) ----
#define S_LEN   2048
#define HID_DIM 2880
#define NH      64
#define NKV     8
#define D_HEAD  64
#define G_GRP   (NH / NKV)
#define SW_WIN  128
#define SCALE_F 0.125f     // 64^-0.5
#define NEG_BIG -1e30f

// GEMM tiling
#define BM 128
#define BN 128
#define BK 32
#define LDK 40             // padded LDS row stride (bf16 elems): 80B, 16B-aligned

typedef __attribute__((ext_vector_type(16))) __bf16 v16bf;
typedef __attribute__((ext_vector_type(8)))  __bf16 v8bf;
typedef __attribute__((ext_vector_type(8)))  float  v8f;

#if defined(__gfx1250__)
#define HAVE_ASYNC_LDS 1
#else
#define HAVE_ASYNC_LDS 0
#endif

__device__ __forceinline__ unsigned short f2bf(float f) {
  unsigned int u = __float_as_uint(f);
  u += 0x7fffu + ((u >> 16) & 1u);      // round-to-nearest-even
  return (unsigned short)(u >> 16);
}

// 16-byte global -> LDS async copy (ASYNCcnt-tracked on gfx1250)
__device__ __forceinline__ void copy16_to_lds(const unsigned short* __restrict__ g,
                                              unsigned short* __restrict__ l) {
#if HAVE_ASYNC_LDS
  // VDST = LDS byte address (flat shared addr truncates to LDS offset, ISA 10.2)
  unsigned int l32 = (unsigned int)(unsigned long long)l;
  asm volatile("global_load_async_to_lds_b128 %0, %1, off"
               :: "v"(l32), "v"(g) : "memory");
#else
  *(uint4*)l = *(const uint4*)g;
#endif
}

__device__ __forceinline__ void wait_copies_keep4() {
#if HAVE_ASYNC_LDS
  asm volatile("s_wait_asynccnt 4" ::: "memory");
#endif
}
__device__ __forceinline__ void wait_copies_all() {
#if HAVE_ASYNC_LDS
  asm volatile("s_wait_asynccnt 0" ::: "memory");
#endif
}

// ---------------- conversion kernels ----------------
__global__ void cvt_f32_to_bf16(const float* __restrict__ in,
                                unsigned short* __restrict__ out, int n) {
  int i = blockIdx.x * blockDim.x + threadIdx.x;
  if (i < n) out[i] = f2bf(in[i]);
}

// in: R x C f32 row-major; out: C x R bf16 row-major (i.e., transposed)
__global__ void transpose_f32_to_bf16(const float* __restrict__ in,
                                      unsigned short* __restrict__ out,
                                      int R, int C) {
  long long i = (long long)blockIdx.x * blockDim.x + threadIdx.x;
  long long total = (long long)R * C;
  if (i >= total) return;
  int c = (int)(i / R);      // output row  (original column)
  int r = (int)(i % R);      // output col  (original row)
  out[i] = f2bf(in[(long long)r * C + c]);
}

// Copies a 128-row x 32-col bf16 tile (row stride = K in global, LDK in LDS).
// 128 rows * 4 chunks(16B) = 512 chunks; 256 threads -> 2 chunks each.
__device__ __forceinline__ void issue_tile_copy(const unsigned short* __restrict__ g,
                                                unsigned short* __restrict__ l,
                                                int K, int tid) {
#pragma unroll
  for (int i = 0; i < 2; ++i) {
    int c   = i * 256 + tid;
    int row = c >> 2;
    int k8  = (c & 3) * 8;
    copy16_to_lds(g + (size_t)row * K + k8, l + row * LDK + k8);
  }
}

// ---------------- bf16 WMMA GEMM, LDS-staged, double-buffered ----------------
// C[M,N] (f32) = A[M,K](bf16 row-major) * BT[N,K]^T (bf16 NxK row-major) + bias[N]
// Block: 256 thr / 8 waves -> 128x128 C tile. Wave: 32x64 (2x4 grid of 16x16).
// BT must be allocated with rows padded to a multiple of BN; stores guarded by n<N.
__global__ __launch_bounds__(256) void gemm_bf16_wmma(
    const unsigned short* __restrict__ A,
    const unsigned short* __restrict__ BT,
    const float* __restrict__ bias,
    float* __restrict__ C,
    int M, int N, int K) {
  __shared__ alignas(16) unsigned short As[2][BM * LDK];
  __shared__ alignas(16) unsigned short Bs[2][BN * LDK];

  const int tid  = threadIdx.x;
  const int lane = tid & 31;
  const int wave = tid >> 5;
  const int l    = lane & 15;
  const int hi   = lane >> 4;

  const int nTilesN = (N + BN - 1) / BN;
  const int m_blk = (blockIdx.x / nTilesN) * BM;
  const int n_blk = (blockIdx.x % nTilesN) * BN;

  const int wm0 = (wave & 3) * 32;   // wave's M offset inside block tile
  const int wn0 = (wave >> 2) * 64;  // wave's N offset inside block tile

  const unsigned short* gA = A  + (size_t)m_blk * K;
  const unsigned short* gB = BT + (size_t)n_blk * K;

  v8f acc[2][4] = {};

  const int ntiles = K / BK;
  // prime the pipeline: tile 0 -> buffer 0
  issue_tile_copy(gA, &As[0][0], K, tid);
  issue_tile_copy(gB, &Bs[0][0], K, tid);

  for (int t = 0; t < ntiles; ++t) {
    const int buf = t & 1;
    if (t + 1 < ntiles) {
      issue_tile_copy(gA + (size_t)(t + 1) * BK, &As[buf ^ 1][0], K, tid);
      issue_tile_copy(gB + (size_t)(t + 1) * BK, &Bs[buf ^ 1][0], K, tid);
      wait_copies_keep4();    // tile t landed; tile t+1 still in flight
    } else {
      wait_copies_all();
    }
    __syncthreads();          // tile t visible to all waves

    const unsigned short* Asb = &As[buf][0];
    const unsigned short* Bsb = &Bs[buf][0];

    // A fragments: lane l holds row; lanes 0-15 K=0..15, 16-31 K=16..31
    v16bf afr[2];
#pragma unroll
    for (int s = 0; s < 2; ++s) {
      const unsigned short* p = Asb + (wm0 + s * 16 + l) * LDK + hi * 16;
      v8bf lo = *(const v8bf*)p;
      v8bf hi8 = *(const v8bf*)(p + 8);
      afr[s] = __builtin_shufflevector(lo, hi8,
          0,1,2,3,4,5,6,7,8,9,10,11,12,13,14,15);
    }
    // B fragments: lane l holds column; lanes 0-15 K=0..15, 16-31 K=16..31
    v16bf bfr[4];
#pragma unroll
    for (int j = 0; j < 4; ++j) {
      const unsigned short* p = Bsb + (wn0 + j * 16 + l) * LDK + hi * 16;
      v8bf lo = *(const v8bf*)p;
      v8bf hi8 = *(const v8bf*)(p + 8);
      bfr[j] = __builtin_shufflevector(lo, hi8,
          0,1,2,3,4,5,6,7,8,9,10,11,12,13,14,15);
    }
#pragma unroll
    for (int s = 0; s < 2; ++s)
#pragma unroll
      for (int j = 0; j < 4; ++j)
        acc[s][j] = __builtin_amdgcn_wmma_f32_16x16x32_bf16(
            false, afr[s], false, bfr[j], (short)0, acc[s][j], false, false);

    __syncthreads();          // all waves done reading buf before it is reused
  }

  // epilogue: D layout VGPR r -> M = r + 8*hi, N = lane&15 within each 16x16
#pragma unroll
  for (int s = 0; s < 2; ++s) {
#pragma unroll
    for (int j = 0; j < 4; ++j) {
      const int n = n_blk + wn0 + j * 16 + l;
      if (n < N) {
        const float bv = bias[n];
#pragma unroll
        for (int r = 0; r < 8; ++r) {
          const int m = m_blk + wm0 + s * 16 + hi * 8 + r;
          C[(size_t)m * N + n] = acc[s][j][r] + bv;
        }
      }
    }
  }
}

// ---------------- RoPE (in place, f32) ----------------
__global__ void rope_kernel(float* __restrict__ x,
                            const float* __restrict__ cosp,
                            const float* __restrict__ sinp, int nheads) {
  int i = blockIdx.x * blockDim.x + threadIdx.x;
  int total = S_LEN * nheads * 32;
  if (i >= total) return;
  int d = i & 31;
  int h = (i >> 5) % nheads;
  int s = i / (nheads * 32);
  float* p = x + (size_t)s * nheads * D_HEAD + (size_t)h * D_HEAD;
  float c0 = cosp[s * D_HEAD + d];
  float s0 = sinp[s * D_HEAD + d];
  float c1 = cosp[s * D_HEAD + d + 32];
  float s1 = sinp[s * D_HEAD + d + 32];
  float x0 = p[d], x1 = p[d + 32];
  p[d]      = x0 * c0 - x1 * s0;
  p[d + 32] = x1 * c1 + x0 * s1;
}

// ---------------- sliding-window attention with sinks ----------------
__global__ void attn_swa_kernel(const float* __restrict__ qf,
                                const float* __restrict__ kf,
                                const float* __restrict__ vf,
                                const float* __restrict__ sinks,
                                unsigned short* __restrict__ attnb) {
  const int lane = threadIdx.x & 31;
  const int wid = blockIdx.x * (blockDim.x >> 5) + (threadIdx.x >> 5);
  if (wid >= NH * S_LEN) return;
  const int qi = wid % S_LEN;
  const int h  = wid / S_LEN;
  const int kv = h / G_GRP;

  const float* qp = qf + (size_t)qi * (NH * D_HEAD) + (size_t)h * D_HEAD;
  const float q0 = qp[lane];
  const float q1 = qp[lane + 32];

  int st = qi - (SW_WIN - 1); if (st < 0) st = 0;
  const int wlen = qi - st + 1;                  // <= 128

  float sc[4];
#pragma unroll
  for (int j = 0; j < 4; ++j) sc[j] = NEG_BIG;

  const float* krow = kf + (size_t)st * (NKV * D_HEAD) + (size_t)kv * D_HEAD;
  for (int tk = 0; tk < wlen; ++tk) {
    const float* kp = krow + (size_t)tk * (NKV * D_HEAD);
    float part = q0 * kp[lane] + q1 * kp[lane + 32];
#pragma unroll
    for (int off = 16; off > 0; off >>= 1)
      part += __shfl_xor(part, off, 32);
    if ((tk & 31) == lane) sc[tk >> 5] = part * SCALE_F;
  }

  const float snk = sinks[h];
  float mx = snk;
#pragma unroll
  for (int j = 0; j < 4; ++j) mx = fmaxf(mx, sc[j]);
#pragma unroll
  for (int off = 16; off > 0; off >>= 1)
    mx = fmaxf(mx, __shfl_xor(mx, off, 32));

  float e[4];
  float dsum = 0.f;
#pragma unroll
  for (int j = 0; j < 4; ++j) {
    int tk = j * 32 + lane;
    e[j] = (tk < wlen) ? __expf(sc[j] - mx) : 0.f;
    dsum += e[j];
  }
#pragma unroll
  for (int off = 16; off > 0; off >>= 1)
    dsum += __shfl_xor(dsum, off, 32);
  dsum += __expf(snk - mx);
  const float inv = 1.f / dsum;

  float a0 = 0.f, a1 = 0.f;
  const float* vrow = vf + (size_t)st * (NKV * D_HEAD) + (size_t)kv * D_HEAD;
  for (int tk = 0; tk < wlen; ++tk) {
    float p = __shfl(e[tk >> 5], tk & 31, 32);
    const float* vp = vrow + (size_t)tk * (NKV * D_HEAD);
    a0 += p * vp[lane];
    a1 += p * vp[lane + 32];
  }

  unsigned short* op = attnb + (size_t)qi * (NH * D_HEAD) + (size_t)h * D_HEAD;
  op[lane]      = f2bf(a0 * inv);
  op[lane + 32] = f2bf(a1 * inv);
}

// ---------------- launcher ----------------
extern "C" void kernel_launch(void* const* d_in, const int* in_sizes, int n_in,
                              void* d_out, int out_size, void* d_ws, size_t ws_size,
                              hipStream_t stream) {
  (void)in_sizes; (void)n_in; (void)out_size; (void)ws_size;
  const float* hs    = (const float*)d_in[0];
  const float* cosp  = (const float*)d_in[1];
  const float* sinp  = (const float*)d_in[2];
  const float* wq    = (const float*)d_in[3];
  const float* bq    = (const float*)d_in[4];
  const float* wk    = (const float*)d_in[5];
  const float* bk    = (const float*)d_in[6];
  const float* wv    = (const float*)d_in[7];
  const float* bv    = (const float*)d_in[8];
  const float* wo    = (const float*)d_in[9];
  const float* bo    = (const float*)d_in[10];
  const float* sinks = (const float*)d_in[11];
  float* out = (float*)d_out;

  const int NQ = NH * D_HEAD;          // 4096
  const int NKVD = NKV * D_HEAD;       // 512
  const int HID_PAD = ((HID_DIM + BN - 1) / BN) * BN;  // 2944

  char* w = (char*)d_ws;
  size_t off = 0;
  auto take = [&](size_t bytes) -> char* {
    char* p = w + off;
    off = (off + bytes + 255) & ~(size_t)255;
    return p;
  };
  unsigned short* Xb    = (unsigned short*)take((size_t)S_LEN * HID_DIM * 2);
  unsigned short* wqT   = (unsigned short*)take((size_t)NQ * HID_DIM * 2);
  unsigned short* wkT   = (unsigned short*)take((size_t)NKVD * HID_DIM * 2);
  unsigned short* wvT   = (unsigned short*)take((size_t)NKVD * HID_DIM * 2);
  unsigned short* woT   = (unsigned short*)take((size_t)HID_PAD * NQ * 2);  // padded rows
  float*          qf    = (float*)take((size_t)S_LEN * NQ * 4);
  float*          kf    = (float*)take((size_t)S_LEN * NKVD * 4);
  float*          vf    = (float*)take((size_t)S_LEN * NKVD * 4);
  unsigned short* attnb = (unsigned short*)take((size_t)S_LEN * NQ * 2);

  // 1) activations -> bf16
  {
    int n = S_LEN * HID_DIM;
    cvt_f32_to_bf16<<<(n + 255) / 256, 256, 0, stream>>>(hs, Xb, n);
  }
  // 2) weights -> transposed bf16 (N x K row-major)
  {
    long long n = (long long)HID_DIM * NQ;
    transpose_f32_to_bf16<<<(unsigned)((n + 255) / 256), 256, 0, stream>>>(
        wq, wqT, HID_DIM, NQ);
  }
  {
    long long n = (long long)HID_DIM * NKVD;
    transpose_f32_to_bf16<<<(unsigned)((n + 255) / 256), 256, 0, stream>>>(
        wk, wkT, HID_DIM, NKVD);
    transpose_f32_to_bf16<<<(unsigned)((n + 255) / 256), 256, 0, stream>>>(
        wv, wvT, HID_DIM, NKVD);
  }
  {
    long long n = (long long)NQ * HID_DIM;
    transpose_f32_to_bf16<<<(unsigned)((n + 255) / 256), 256, 0, stream>>>(
        wo, woT, NQ, HID_DIM);
  }

  auto launch_gemm = [&](const unsigned short* A, const unsigned short* BT,
                         const float* bias, float* Cp, int M, int N, int K) {
    int blocks = (M / BM) * ((N + BN - 1) / BN);
    gemm_bf16_wmma<<<blocks, 256, 0, stream>>>(A, BT, bias, Cp, M, N, K);
  };

  // 3) Q/K/V projections (bf16 WMMA, f32 accumulate, LDS double-buffered)
  launch_gemm(Xb, wqT, bq, qf, S_LEN, NQ,   HID_DIM);
  launch_gemm(Xb, wkT, bk, kf, S_LEN, NKVD, HID_DIM);
  launch_gemm(Xb, wvT, bv, vf, S_LEN, NKVD, HID_DIM);

  // 4) RoPE on q and k
  {
    int n = S_LEN * NH * 32;
    rope_kernel<<<(n + 255) / 256, 256, 0, stream>>>(qf, cosp, sinp, NH);
  }
  {
    int n = S_LEN * NKV * 32;
    rope_kernel<<<(n + 255) / 256, 256, 0, stream>>>(kf, cosp, sinp, NKV);
  }

  // 5) sliding-window attention with sinks -> bf16 activations
  {
    int waves = NH * S_LEN;
    attn_swa_kernel<<<(waves + 7) / 8, 256, 0, stream>>>(qf, kf, vf, sinks, attnb);
  }

  // 6) output projection (bf16 WMMA) -> f32 result (N=2880, padded BT rows)
  launch_gemm(attnb, woT, bo, out, S_LEN, HID_DIM, NQ);
}
